// TimeMixing_2018634629630
// MI455X (gfx1250) — matrix operand order repaired and checked by reading, not compiled, not run on hardware
//
#include <hip/hip_runtime.h>
#include <hip/hip_bf16.h>

// ---------------------------------------------------------------------------
// RWKV7-style time mixing for MI455X (gfx1250, wave32, WMMA, async-to-LDS).
//   B=4, L=2048, D=1024, H=16, HD=64, K=4
// Pipeline:
//   1) f32 -> bf16 converts (x, 8 weights)
//   2) 7 fused projection GEMMs: bf16 WMMA 16x16x32, f32 accum, fused act,
//      double-buffered LDS tiles fed by GLOBAL_LOAD_ASYNC_TO_LDS_B128
//   3) sequential scan: 64 blocks (B*H), 64 lanes, state row per lane in VGPRs,
//      column vectors async-loaded straight into LDS
//   4) gate (g * out -> bf16), Wo GEMM (WMMA), conv+RMSnorm epilogue
// ---------------------------------------------------------------------------

#define Bc  4
#define Lc  2048
#define Dc  1024
#define Hc  16
#define HDc 64
#define Kc  4
#define Mc  (Bc * Lc)            // 8192 rows

typedef __attribute__((ext_vector_type(16))) __bf16 v16bf;
typedef __attribute__((ext_vector_type(8)))  float  v8f;
typedef __attribute__((ext_vector_type(4)))  __bf16 v4bf;
typedef int v4i_vs __attribute__((vector_size(16)));   // builtin's b128 elt type

#define AS1 __attribute__((address_space(1)))
#define AS3 __attribute__((address_space(3)))

// ----------------------- async global -> LDS helpers -----------------------
__device__ __forceinline__ void async_copy16(const __bf16* g, __bf16* l) {
#if __has_builtin(__builtin_amdgcn_global_load_async_to_lds_b128)
    __builtin_amdgcn_global_load_async_to_lds_b128(
        (AS1 v4i_vs*)g, (AS3 v4i_vs*)l, 0, 0);
#else
    *(uint4*)l = *(const uint4*)g;
#endif
}

__device__ __forceinline__ void async_copy4(const float* g, float* l) {
#if __has_builtin(__builtin_amdgcn_global_load_async_to_lds_b32)
    __builtin_amdgcn_global_load_async_to_lds_b32(
        (AS1 int*)g, (AS3 int*)l, 0, 0);
#else
    *l = *g;
#endif
}

__device__ __forceinline__ void async_wait0() {
#if __has_builtin(__builtin_amdgcn_s_wait_asynccnt)
    __builtin_amdgcn_s_wait_asynccnt(0);
#elif __has_builtin(__builtin_amdgcn_global_load_async_to_lds_b32)
    asm volatile("s_wait_asynccnt 0x0" ::: "memory");
#endif
}

// ----------------------------- converts ------------------------------------
__global__ void f32_to_bf16_vec4(const float* __restrict__ in,
                                 __bf16* __restrict__ out, int n4) {
    int i = blockIdx.x * blockDim.x + threadIdx.x;
    if (i >= n4) return;
    float4 f = ((const float4*)in)[i];
    v4bf o;
    o[0] = (__bf16)f.x; o[1] = (__bf16)f.y; o[2] = (__bf16)f.z; o[3] = (__bf16)f.w;
    ((v4bf*)out)[i] = o;
}

// ------------------------------ WMMA GEMM ----------------------------------
// Out[m,n] = act( sum_k A[m,k] * W[widx][n,k] )   (nn.Linear: y = x @ W^T)
// Tile: BM=128, BN=128, BK=32. 256 threads = 8 waves; wave = 32x64 subtile.
// Double-buffered LDS; tiles streamed in with async-to-LDS (ASYNCcnt).
#define BM 128
#define BN 128
#define BK 32

__device__ __forceinline__ float act_apply(int act, float v, float bdv) {
    if (act == 1) return 1.0f / (1.0f + __expf(-v));          // sigmoid
    if (act == 2) {                                           // decay gate
        float w = __expf(-bdv) * __expf(-__expf(v));
        return fminf(fmaxf(w, 0.01f), 0.99f);
    }
    return v;
}

union FragBF { v16bf v; uint4 q[2]; };

__global__ __launch_bounds__(256)
void wmma_gemm_kernel(const __bf16* __restrict__ A,      // M x 1024 bf16
                      const __bf16* __restrict__ Wall,   // 8 x (1024x1024) bf16
                      float* __restrict__ Out,           // per-z: M x 1024 f32
                      const float* __restrict__ bd,      // base_decay (act==2)
                      int M, int widx_base, int act_override,
                      long outStrideMat) {
    __shared__ __align__(16) __bf16 As[2][BM * BK];
    __shared__ __align__(16) __bf16 Bs[2][BN * BK];

    const int z    = blockIdx.z;
    const int widx = widx_base + z;
    // projection order: r,k,v,g,w,a,kk -> sigmoid,id,id,sigmoid,decay,id,id
    const int actTab[7] = {1, 0, 0, 1, 2, 0, 0};
    const int act = (act_override >= 0) ? act_override : actTab[z];

    const int tid  = threadIdx.x;
    const int lane = tid & 31;
    const int wave = tid >> 5;
    const int wm   = wave & 3;     // 0..3 -> 32-row stripes
    const int wn   = wave >> 2;    // 0..1 -> 64-col stripes
    const int gm0  = blockIdx.x * BM;
    const int gn0  = blockIdx.y * BN;

    const __bf16* Wmat = Wall + (long)widx * Dc * Dc;

    // per-thread tile-chunk coordinates (2 x 16B chunks for A, 2 for B)
    const int ch0r = tid >> 2,          ch0c = (tid & 3) * 8;
    const int ch1r = (tid + 256) >> 2,  ch1c = ((tid + 256) & 3) * 8;

    v8f c[2][4];
    #pragma unroll
    for (int mi = 0; mi < 2; ++mi)
        #pragma unroll
        for (int ni = 0; ni < 4; ++ni)
            c[mi][ni] = (v8f)(0.0f);

    const int halfsel = lane >> 4;     // 0 or 1
    const int l15     = lane & 15;

    // prologue: stage first K-tile into buffer 0
    async_copy16(&A[(long)(gm0 + ch0r) * Dc + ch0c],    &As[0][ch0r * BK + ch0c]);
    async_copy16(&Wmat[(long)(gn0 + ch0r) * Dc + ch0c], &Bs[0][ch0r * BK + ch0c]);
    async_copy16(&A[(long)(gm0 + ch1r) * Dc + ch1c],    &As[0][ch1r * BK + ch1c]);
    async_copy16(&Wmat[(long)(gn0 + ch1r) * Dc + ch1c], &Bs[0][ch1r * BK + ch1c]);
    async_wait0();
    __syncthreads();

    int cur = 0;
    for (int kb = 0; kb < Dc; kb += BK) {
        // prefetch next K-tile into the other buffer while computing this one
        if (kb + BK < Dc) {
            const int nxt = cur ^ 1;
            const int kn  = kb + BK;
            async_copy16(&A[(long)(gm0 + ch0r) * Dc + kn + ch0c],
                         &As[nxt][ch0r * BK + ch0c]);
            async_copy16(&Wmat[(long)(gn0 + ch0r) * Dc + kn + ch0c],
                         &Bs[nxt][ch0r * BK + ch0c]);
            async_copy16(&A[(long)(gm0 + ch1r) * Dc + kn + ch1c],
                         &As[nxt][ch1r * BK + ch1c]);
            async_copy16(&Wmat[(long)(gn0 + ch1r) * Dc + kn + ch1c],
                         &Bs[nxt][ch1r * BK + ch1c]);
        }

        FragBF a[2], b[4];
        #pragma unroll
        for (int mi = 0; mi < 2; ++mi) {
            int m = wm * 32 + mi * 16 + l15;
            // A 16x32 bf16 layout: lane holds K = off..off+7 and 16+off..16+off+7
            a[mi].q[0] = *(const uint4*)&As[cur][m * BK + halfsel * 8];
            a[mi].q[1] = *(const uint4*)&As[cur][m * BK + 16 + halfsel * 8];
        }
        #pragma unroll
        for (int ni = 0; ni < 4; ++ni) {
            int n = wn * 64 + ni * 16 + l15;
            // B 32x16 bf16 layout: lane holds K = off..off+7 and 16+off..16+off+7
            b[ni].q[0] = *(const uint4*)&Bs[cur][n * BK + halfsel * 8];
            b[ni].q[1] = *(const uint4*)&Bs[cur][n * BK + 16 + halfsel * 8];
        }
        #pragma unroll
        for (int mi = 0; mi < 2; ++mi)
            #pragma unroll
            for (int ni = 0; ni < 4; ++ni)
                c[mi][ni] = __builtin_amdgcn_wmma_f32_16x16x32_bf16(
                    false, a[mi].v, false, b[ni].v,
                    (short)0, c[mi][ni], false, false);

        async_wait0();      // my async writes into 'nxt' are done
        __syncthreads();    // everyone's are done; 'cur' reads also complete
        cur ^= 1;
    }

    float* Outz = Out + (long)z * outStrideMat;
    #pragma unroll
    for (int mi = 0; mi < 2; ++mi) {
        #pragma unroll
        for (int ni = 0; ni < 4; ++ni) {
            int rowBase = gm0 + wm * 32 + mi * 16 + (halfsel << 3);
            int col     = gn0 + wn * 64 + ni * 16 + l15;
            float bdv   = (act == 2) ? bd[col] : 0.0f;
            #pragma unroll
            for (int v = 0; v < 8; ++v) {
                Outz[(long)(rowBase + v) * Dc + col] =
                    act_apply(act, c[mi][ni][v], bdv);
            }
        }
    }
}

// ------------------------------- scan --------------------------------------
// One block per (b,h); 64 lanes; lane i owns state row S[i][0..63] in VGPRs.
// Column vectors (v, kk, a) are async-loaded straight into LDS.
__global__ __launch_bounds__(64)
void scan_kernel(const float* __restrict__ proj,   // 7 mats of M*1024 f32
                 float* __restrict__ scan_out,     // M*1024 f32
                 float* __restrict__ state_out) {  // B*H*64*64 f32
    const int bh = blockIdx.x;
    const int b  = bh >> 4;
    const int h  = bh & 15;
    const int i  = threadIdx.x;

    const long MN = (long)Mc * Dc;
    const float* R  = proj + 0 * MN;
    const float* Kp = proj + 1 * MN;
    const float* V  = proj + 2 * MN;
    const float* W  = proj + 4 * MN;
    const float* Aa = proj + 5 * MN;
    const float* KK = proj + 6 * MN;

    __shared__ float vS[64], kkS[64], aS[64];

    float S[64];
    #pragma unroll
    for (int j = 0; j < 64; ++j) S[j] = 0.0f;

    long idx = ((long)b * Lc) * Dc + h * HDc + i;
    for (int t = 0; t < Lc; ++t, idx += Dc) {
        // column vectors -> LDS without a VGPR round-trip (ASYNCcnt path)
        async_copy4(&V[idx],  &vS[i]);
        async_copy4(&KK[idx], &kkS[i]);
        async_copy4(&Aa[idx], &aS[i]);

        float r_i = R[idx];
        float k_i = Kp[idx];
        float w_i = W[idx];

        if (t + 1 < Lc) {   // hide next-step latency across the serial chain
            __builtin_prefetch(&R[idx + Dc],  0, 1);
            __builtin_prefetch(&Kp[idx + Dc], 0, 1);
            __builtin_prefetch(&V[idx + Dc],  0, 1);
            __builtin_prefetch(&W[idx + Dc],  0, 1);
            __builtin_prefetch(&Aa[idx + Dc], 0, 1);
            __builtin_prefetch(&KK[idx + Dc], 0, 1);
        }

        async_wait0();
        __syncthreads();

        // sa_i = sum_j S[i][j] * (-kk_j)
        float sa = 0.0f;
        #pragma unroll
        for (int j = 0; j < 64; ++j) sa -= S[j] * kkS[j];

        // S[i][j] = w_i*S + sa*(kk_j*a_j) + k_i*v_j ; row sum of squares
        float ss = 0.0f;
        #pragma unroll
        for (int j = 0; j < 64; ++j) {
            float ka = kkS[j] * aS[j];
            S[j] = fmaf(w_i, S[j], fmaf(sa, ka, k_i * vS[j]));
            ss  = fmaf(S[j], S[j], ss);
        }
        float inv = 1.0f / (sqrtf(ss * (1.0f / 64.0f)) + 1e-5f);

        float rs = 0.0f;
        #pragma unroll
        for (int j = 0; j < 64; ++j) {
            S[j] *= inv;
            rs   += S[j];
        }
        scan_out[idx] = r_i * rs;   // o_i = r_i * sum_j S[i][j]
        __syncthreads();            // protect LDS before next step's writes
    }

    float* so = state_out + ((long)(b * Hc + h) * HDc + i) * HDc;
    #pragma unroll
    for (int j = 0; j < 64; ++j) so[j] = S[j];
}

// ------------------------------- gate --------------------------------------
__global__ void gate_kernel(const float* __restrict__ g,
                            const float* __restrict__ so,
                            __bf16* __restrict__ gated, int n4) {
    int i = blockIdx.x * blockDim.x + threadIdx.x;
    if (i >= n4) return;
    float4 gg = ((const float4*)g)[i];
    float4 oo = ((const float4*)so)[i];
    v4bf o;
    o[0] = (__bf16)(gg.x * oo.x); o[1] = (__bf16)(gg.y * oo.y);
    o[2] = (__bf16)(gg.z * oo.z); o[3] = (__bf16)(gg.w * oo.w);
    ((v4bf*)gated)[i] = o;
}

// ----------------------- conv + RMSnorm epilogue ---------------------------
// One block per (b,l) row of 1024; 256 threads, 4 columns each.
__global__ __launch_bounds__(256)
void final_kernel(const float* __restrict__ x,
                  const float* __restrict__ y2,
                  const float* __restrict__ conv_w,  // (D,1,K) flat
                  const float* __restrict__ lnw,
                  float* __restrict__ out) {
    const int row = blockIdx.x;            // b*L + l
    const int b   = row / Lc;
    const int l   = row - b * Lc;
    const int tid = threadIdx.x;

    __shared__ float red[256];
    const long rbase = (long)row * Dc;

    float local = 0.0f;
    #pragma unroll
    for (int q = 0; q < 4; ++q) {
        float yv = y2[rbase + tid + q * 256];
        local = fmaf(yv, yv, local);
    }
    red[tid] = local;
    __syncthreads();
    #pragma unroll
    for (int s = 128; s > 0; s >>= 1) {
        if (tid < s) red[tid] += red[tid + s];
        __syncthreads();
    }
    const float inv = 1.0f / (sqrtf(red[0] * (1.0f / (float)Dc)) + 1e-6f);

    #pragma unroll
    for (int q = 0; q < 4; ++q) {
        int d = tid + q * 256;
        // depthwise causal conv: xc[l] = sum_k x[l+k-3] * w[k]
        float xc = 0.0f;
        #pragma unroll
        for (int kk = 0; kk < Kc; ++kk) {
            int ls = l + kk - (Kc - 1);
            if (ls >= 0 && ls < Lc)
                xc = fmaf(x[((long)b * Lc + ls) * Dc + d], conv_w[d * Kc + kk], xc);
        }
        float xv = x[rbase + d];
        float yv = y2[rbase + d];
        out[rbase + d] = xv + 0.1f * xc + yv * inv * lnw[d];
    }
}

// ------------------------------ launch -------------------------------------
extern "C" void kernel_launch(void* const* d_in, const int* in_sizes, int n_in,
                              void* d_out, int out_size, void* d_ws, size_t ws_size,
                              hipStream_t stream) {
    (void)in_sizes; (void)n_in; (void)out_size; (void)ws_size;

    const float* x      = (const float*)d_in[0];
    const float* conv_w = (const float*)d_in[9];
    const float* bd     = (const float*)d_in[10];
    const float* lnw    = (const float*)d_in[11];
    float*       out    = (float*)d_out;

    char* ws = (char*)d_ws;
    const long nX  = (long)Mc * Dc;          // 8,388,608
    const long nW  = (long)Dc * Dc;          // 1,048,576

    __bf16* xb    = (__bf16*)(ws);                                   // 16 MB
    __bf16* Wball = (__bf16*)(ws + nX * 2);                          // 16 MB
    float*  proj  = (float*) (ws + nX * 2 + 8 * nW * 2);             // 224 MB
    float*  sout  = (float*) (ws + nX * 2 + 8 * nW * 2 + 7 * nX * 4);// 32 MB
    __bf16* gated = xb;            // reuse: xb dead after projection GEMMs
    float*  y2    = proj;          // reuse: 'r' slot dead after scan

    const int cvtB = 256;

    // 1) converts
    f32_to_bf16_vec4<<<(int)((nX / 4 + cvtB - 1) / cvtB), cvtB, 0, stream>>>(
        x, xb, (int)(nX / 4));
    for (int i = 0; i < 8; ++i) {
        f32_to_bf16_vec4<<<(int)((nW / 4 + cvtB - 1) / cvtB), cvtB, 0, stream>>>(
            (const float*)d_in[1 + i], Wball + (long)i * nW, (int)(nW / 4));
    }

    // 2) 7 projection GEMMs (z selects weight + activation)
    dim3 ggrid(Mc / BM, Dc / BN, 7);
    wmma_gemm_kernel<<<ggrid, 256, 0, stream>>>(
        xb, Wball, proj, bd, Mc, /*widx_base=*/0, /*act_override=*/-1,
        (long)Mc * Dc);

    // 3) scan; final state written straight into d_out tail
    float* state_out = out + (long)Mc * Dc;
    scan_kernel<<<Bc * Hc, 64, 0, stream>>>(proj, sout, state_out);

    // 4) gate -> bf16
    gate_kernel<<<(int)((nX / 4 + cvtB - 1) / cvtB), cvtB, 0, stream>>>(
        proj + 3 * nX, sout, gated, (int)(nX / 4));

    // 5) output projection (Wo = weight index 7, identity activation)
    dim3 ogrid(Mc / BM, Dc / BN, 1);
    wmma_gemm_kernel<<<ogrid, 256, 0, stream>>>(
        gated, Wball, y2, bd, Mc, /*widx_base=*/7, /*act_override=*/0,
        (long)Mc * Dc);

    // 6) conv + RMSnorm epilogue
    final_kernel<<<Mc, 256, 0, stream>>>(x, y2, conv_w, lnw, out);
}
